// Time_sync_74981539053784
// MI455X (gfx1250) — matrix operand order repaired and checked by reading, not compiled, not run on hardware
//
#include <hip/hip_runtime.h>
#include <hip/hip_bf16.h>
#include <math.h>

// ---------------------------------------------------------------------------
// CDNA5 (gfx1250) wave32 WMMA bf16 GEMM pipeline for the TDNN reference.
//  - activations bf16, f32 WMMA accumulate (v_wmma_f32_16x16x32_bf16)
//  - A tile staged via GLOBAL_LOAD_ASYNC_TO_LDS_B128 (ASYNCcnt)
//  - W tile staged via TENSOR_LOAD_TO_LDS 2D descriptor (TENSORcnt, TDM)
//  - double-buffered LDS, one workgroup barrier per K-step
// ---------------------------------------------------------------------------

typedef __attribute__((ext_vector_type(16))) __bf16 v16bf;
typedef __attribute__((ext_vector_type(8)))  float  v8f;
typedef __attribute__((ext_vector_type(4)))  unsigned u32x4;
typedef __attribute__((ext_vector_type(8)))  unsigned u32x8;

#define BM 128           // rows (time steps) per block
#define BN 64            // output channels per block
#define BK 32            // K per WMMA step
#define LDSS 48          // LDS row stride in bf16 elems (96B, 16B-aligned)

__device__ __forceinline__ unsigned short f2bf(float f) {
  unsigned u = __float_as_uint(f);
  u += 0x7FFFu + ((u >> 16) & 1u);          // round-to-nearest-even
  return (unsigned short)(u >> 16);
}
__device__ __forceinline__ float bf2f(unsigned short h) {
  return __uint_as_float((unsigned)h << 16);
}

// One 16B async global->LDS transfer per active lane (ASYNCcnt tracked).
__device__ __forceinline__ void async_b128(unsigned lds_off, unsigned gvoff,
                                           const void* base) {
  asm volatile("global_load_async_to_lds_b128 %0, %1, %2"
               :: "v"(lds_off), "v"(gvoff), "s"(base) : "memory");
}
__device__ __forceinline__ void wait_async0() {
  asm volatile("s_wait_asynccnt 0" ::: "memory");
}

// TDM: load a 2D tile (tile_d1 rows x tile_d0 elems of 2B) from global
// (row stride stride_d0 elems) into LDS at lds_addr, padding each row of
// 16 DWORDs with 8 DWORDs so the LDS row stride is 96B (== LDSS elems).
// Descriptor layout per CDNA5 ISA 8.3/8.4 (groups 0 and 1, 2D, no cluster).
__device__ __forceinline__ void tdm_load_2d(unsigned lds_addr,
                                            const void* gtile,
                                            unsigned tile_d0, unsigned tile_d1,
                                            unsigned stride_d0) {
  unsigned long long ga = (unsigned long long)(size_t)gtile;
  u32x4 g0;
  g0[0] = 1u;                                        // count=1 (valid user D#)
  g0[1] = lds_addr;                                  // lds_addr (bytes)
  g0[2] = (unsigned)(ga & 0xFFFFFFFFu);              // global_addr[31:0]
  g0[3] = (unsigned)((ga >> 32) & 0x01FFFFFFu) | (2u << 30); // [56:32]|type=2
  u32x8 g1;
  g1[0] = (1u << 16)            // data_size = 1 -> 2 bytes
        | (1u << 20)            // pad_enable
        | (3u << 22)            // pad_interval code 3 -> every 16 DWORDs (64B)
        | (7u << 25);           // pad_amount  code 7 -> 8 DWORDs (32B)
  g1[1] = (tile_d0 & 0xFFFFu) << 16;                 // tensor_dim0 = tile_d0
  g1[2] = (tile_d1 & 0xFFFFu) << 16;                 // tensor_dim1 = tile_d1
  g1[3] = (tile_d0 & 0xFFFFu) << 16;                 // tile_dim0
  g1[4] = (tile_d1 & 0xFFFFu);                       // tile_dim1 (tile_dim2=0)
  g1[5] = stride_d0;                                 // tensor_dim0_stride[31:0]
  g1[6] = 0u;                                        // stride[47:32], d1_stride lo
  g1[7] = 0u;                                        // d1_stride hi (unused, 2D)
  asm volatile("tensor_load_to_lds %0, %1" :: "s"(g0), "s"(g1) : "memory");
}

// Generic: out[b*T_out+t][n] = act( sum_taps A[b*T_in + t + off_tap] . W_tap[n]
//                                   + bias[n] (+ resid + rbias[n]) )
__global__ __launch_bounds__(256)
void gemm_wmma_bf16(const void* __restrict__ Aptr, int a_is_f32,
                    int T_in, int T_out, int K, int N,
                    int off0, int off1, int ntaps,
                    const unsigned short* __restrict__ W0,
                    const unsigned short* __restrict__ W1,
                    const float* __restrict__ bias,
                    const unsigned short* __restrict__ resid,
                    const float* __restrict__ rbias,
                    int act,                       // 0=none 1=relu 2=tanh
                    unsigned short* __restrict__ out_bf,
                    float* __restrict__ out_f32) {
  __shared__ __align__(16) unsigned short As[2][BM * LDSS];
  __shared__ __align__(16) unsigned short Ws[2][BN * LDSS];

  const int tid  = threadIdx.x;
  const int lane = tid & 31;
  const int wave = tid >> 5;                 // 0..7 -> M sub-tile
  const int b    = blockIdx.z;
  const int t0   = blockIdx.y * BM;
  const int n0   = blockIdx.x * BN;

  v8f c[4];
  for (int i = 0; i < 4; ++i)
    for (int j = 0; j < 8; ++j) c[i][j] = 0.f;

  // staging roles
  const int arow = tid >> 1;                 // 0..127 (A row, 32B each)
  const int aseg = tid & 1;                  // 16 elems each
  const int wrow = tid >> 2;                 // 0..63  (W row, 16B each)
  const int wseg = tid & 3;                  // 8 elems each
  const int t_a  = t0 + arow;
  const long a_row_base = (long)b * T_in;

  const int kSteps = (K + BK - 1) / BK;
  const int total  = ntaps * kSteps;
  const bool aRowsFull = (t0 + BM <= T_out);
  const bool kAligned  = ((K & 7) == 0);
  const bool wColsFull = (n0 + BN <= N);

  // Stage tile for flattened step s into LDS buffer bi.
  auto stage = [&](int s, int bi) {
    const int tap = (s >= kSteps) ? 1 : 0;
    const int k0  = (s - tap * kSteps) * BK;
    const int off = tap ? off1 : off0;
    const unsigned short* W = tap ? W1 : W0;
    // ---- A tile (BM x BK): per-lane async global->LDS ----
    {
      const int kk = k0 + aseg * 16;
      unsigned short* dst = &As[bi][arow * LDSS + aseg * 16];
      const bool rv = (t_a < T_out);
      if (!a_is_f32 && kAligned && aRowsFull && (kk + 16 <= K)) {
        const unsigned lds  = (unsigned)(size_t)dst;
        const unsigned voff =
            (unsigned)(((a_row_base + t_a + off) * (long)K + kk) * 2);
        async_b128(lds,      voff,      Aptr);
        async_b128(lds + 16, voff + 16, Aptr);
      } else if (!a_is_f32) {
        const unsigned short* src = (const unsigned short*)Aptr +
            (a_row_base + t_a + off) * (long)K + kk;
        if (rv && (kk + 16 <= K)) {
          ((uint4*)dst)[0] = ((const uint4*)src)[0];
          ((uint4*)dst)[1] = ((const uint4*)src)[1];
        } else {
          for (int e = 0; e < 16; ++e)
            dst[e] = (rv && kk + e < K) ? src[e] : (unsigned short)0;
        }
      } else {
        const float* src = (const float*)Aptr +
            (a_row_base + t_a + off) * (long)K + kk;
        if (rv && (kk + 16 <= K)) {
          for (int v = 0; v < 4; ++v) {
            float4 f = ((const float4*)src)[v];
            dst[v * 4 + 0] = f2bf(f.x);
            dst[v * 4 + 1] = f2bf(f.y);
            dst[v * 4 + 2] = f2bf(f.z);
            dst[v * 4 + 3] = f2bf(f.w);
          }
        } else {
          for (int e = 0; e < 16; ++e)
            dst[e] = (rv && kk + e < K) ? f2bf(src[e]) : (unsigned short)0;
        }
      }
    }
    // ---- W tile (BN x BK), W is [N,K] row-major bf16 ----
    if (kAligned && wColsFull && (k0 + BK <= K)) {
      // full tile: one TDM 2D load for the whole block, issued by wave 0
      if (wave == 0) {
        tdm_load_2d((unsigned)(size_t)&Ws[bi][0],
                    (const void*)(W + (long)n0 * K + k0),
                    /*tile_d0=*/BK, /*tile_d1=*/BN,
                    /*stride_d0=*/(unsigned)K);
      }
    } else {
      const int kk = k0 + wseg * 8;
      const int nn = n0 + wrow;
      unsigned short* dst = &Ws[bi][wrow * LDSS + wseg * 8];
      if (nn < N && (kk + 8 <= K)) {
        const unsigned short* src = W + (long)nn * K + kk;
        uint2 u0 = ((const uint2*)src)[0];
        uint2 u1 = ((const uint2*)src)[1];
        uint4 u; u.x = u0.x; u.y = u0.y; u.z = u1.x; u.w = u1.y;
        *((uint4*)dst) = u;
      } else {
        const unsigned short* src = W + (long)nn * K + kk;
        for (int e = 0; e < 8; ++e)
          dst[e] = (nn < N && kk + e < K) ? src[e] : (unsigned short)0;
      }
    }
  };

  // prologue: stage step 0 into buffer 0
  stage(0, 0);
  wait_async0();
  __builtin_amdgcn_s_wait_tensorcnt(0);
  __syncthreads();

  int cur = 0;
  for (int s = 0; s < total; ++s) {
    // stage next tile into the other buffer (overlaps with WMMA below)
    if (s + 1 < total) stage(s + 1, cur ^ 1);

    // ---- WMMA compute on buffer `cur`: 1 A-frag, 4 B-frags, 4 wmma ----
    {
      const int m  = lane & 15;
      const int hf = lane >> 4;
      // A 16x32 bf16 layout: lanes0-15 hold K0..7/K16..23, lanes16-31
      // hold K8..15/K24..31 (per-row).
      union { v16bf v; uint4 q[2]; } af;
      const unsigned short* ab = &As[cur][(wave * 16 + m) * LDSS];
      af.q[0] = *(const uint4*)(ab + hf * 8);
      af.q[1] = *(const uint4*)(ab + 16 + hf * 8);
#pragma unroll
      for (int ni = 0; ni < 4; ++ni) {
        // B 32x16 bf16 layout: lanes0-15 col n hold K0..7/K16..23, lanes16-31 K8..15/K24..31.
        union { v16bf v; uint4 q[2]; } bfm;
        const unsigned short* bb = &Ws[cur][(ni * 16 + m) * LDSS];
        bfm.q[0] = *(const uint4*)(bb + hf * 8);
        bfm.q[1] = *(const uint4*)(bb + 16 + hf * 8);
        c[ni] = __builtin_amdgcn_wmma_f32_16x16x32_bf16(
            false, af.v, false, bfm.v, (short)0, c[ni], false, false);
      }
    }

    wait_async0();                          // my async A stage is complete
    __builtin_amdgcn_s_wait_tensorcnt(0);   // my TDM W stage is complete
    __syncthreads();                        // all stages + reads of `cur` done
    cur ^= 1;
  }

  // ---- store: D f32 16x16 layout: vgpr r -> M = r + 8*(lane>=16), N = lane%16
  {
    const int m16 = lane & 15;
    const int hf  = lane >> 4;
#pragma unroll
    for (int ni = 0; ni < 4; ++ni) {
      const int col = n0 + ni * 16 + m16;
      if (col >= N) continue;
      const float bcol = bias ? bias[col] : 0.f;
      const float rb   = rbias ? rbias[col] : 0.f;
#pragma unroll
      for (int r = 0; r < 8; ++r) {
        const int t = t0 + wave * 16 + r + hf * 8;
        if (t >= T_out) continue;
        const long orow = (long)b * T_out + t;
        float v = c[ni][r] + bcol;
        if (resid) v += bf2f(resid[orow * (long)N + col]) + rb;
        if (act == 1)      v = fmaxf(v, 0.f);
        else if (act == 2) v = tanhf(v);
        if (out_bf)  out_bf[orow * (long)N + col]  = f2bf(v);
        if (out_f32) out_f32[orow * (long)N + col] = v;
      }
    }
  }
}

// ---------------------------------------------------------------------------
__global__ void k_f32_to_bf16(const float* __restrict__ s,
                              unsigned short* __restrict__ d, long n) {
  long i = (long)blockIdx.x * blockDim.x + threadIdx.x;
  if (i < n) d[i] = f2bf(s[i]);
}

// kernel[o][i][k] (k=0,1) -> w0[o][i], w1[o][i]
__global__ void k_tdnn_split(const float* __restrict__ k,
                             unsigned short* __restrict__ w0,
                             unsigned short* __restrict__ w1, int OI) {
  int i = blockIdx.x * blockDim.x + threadIdx.x;
  if (i < OI) {
    w0[i] = f2bf(k[2 * i + 0]);
    w1[i] = f2bf(k[2 * i + 1]);
  }
}

// softmax over t for fixed (b, j); scores [b, t, 5] -> At [b, j, t]
__global__ __launch_bounds__(256)
void k_softmax_t(const float* __restrict__ scores, float* __restrict__ At,
                 int T4) {
  const int bj = blockIdx.x;
  const int b = bj / 5, j = bj % 5;
  const float* s = scores + (long)b * T4 * 5 + j;
  __shared__ float red[256];
  float mx = -1e30f;
  for (int t = threadIdx.x; t < T4; t += 256) mx = fmaxf(mx, s[(long)t * 5]);
  red[threadIdx.x] = mx; __syncthreads();
  for (int st = 128; st > 0; st >>= 1) {
    if (threadIdx.x < st) red[threadIdx.x] = fmaxf(red[threadIdx.x], red[threadIdx.x + st]);
    __syncthreads();
  }
  mx = red[0]; __syncthreads();
  float sum = 0.f;
  for (int t = threadIdx.x; t < T4; t += 256) sum += __expf(s[(long)t * 5] - mx);
  red[threadIdx.x] = sum; __syncthreads();
  for (int st = 128; st > 0; st >>= 1) {
    if (threadIdx.x < st) red[threadIdx.x] += red[threadIdx.x + st];
    __syncthreads();
  }
  sum = red[0];
  const float inv = 1.f / sum;
  float* o = At + ((long)b * 5 + j) * T4;
  for (int t = threadIdx.x; t < T4; t += 256) o[t] = __expf(s[(long)t * 5] - mx) * inv;
}

// pooled[b][j][d] = sum_t At[b,j,t] * h[b,t,d]   (h bf16, d=0..127)
__global__ void k_pool(const float* __restrict__ At,
                       const unsigned short* __restrict__ h,
                       float* __restrict__ pooled, int T4) {
  const int j = blockIdx.x, b = blockIdx.y, d = threadIdx.x;
  const float* a = At + ((long)b * 5 + j) * T4;
  const unsigned short* hb = h + (long)b * T4 * 128 + d;
  float acc = 0.f;
  for (int t = 0; t < T4; ++t) acc += a[t] * bf2f(hb[(long)t * 128]);
  pooled[((long)b * 5 + j) * 128 + d] = acc;
}

// out[b][o] = relu( pooled[b][:640] . bnw_w[o][:640] + bnw_b[o] )
__global__ void k_final(const float* __restrict__ pooled,
                        const float* __restrict__ w,
                        const float* __restrict__ bias,
                        float* __restrict__ out) {
  const int b = blockIdx.x, o = threadIdx.x;
  const float* p = pooled + (long)b * 640;
  const float* wr = w + (long)o * 640;
  float acc = bias[o];
  for (int k = 0; k < 640; ++k) acc += p[k] * wr[k];
  out[(long)b * 128 + o] = fmaxf(acc, 0.f);
}

// ---------------------------------------------------------------------------
extern "C" void kernel_launch(void* const* d_in, const int* in_sizes, int n_in,
                              void* d_out, int out_size, void* d_ws, size_t ws_size,
                              hipStream_t stream) {
  (void)in_sizes; (void)n_in; (void)out_size; (void)ws_size;
  const int B = 64, T1 = 800, D = 500, H = 512;
  const int T2 = 797, T3 = 791, T4 = 782;

  const float* x     = (const float*)d_in[0];
  const float* fc1_w = (const float*)d_in[1];
  const float* fc1_b = (const float*)d_in[2];
  const float *rw1[4], *rb1[4], *rw2[4], *rb2[4], *rbs[4];
  for (int i = 0; i < 4; ++i) {
    rw1[i] = (const float*)d_in[3 + 5 * i];
    rb1[i] = (const float*)d_in[4 + 5 * i];
    rw2[i] = (const float*)d_in[5 + 5 * i];
    rb2[i] = (const float*)d_in[6 + 5 * i];
    rbs[i] = (const float*)d_in[7 + 5 * i];
  }
  const float *tdk[3], *tdb[3];
  for (int i = 0; i < 3; ++i) {
    tdk[i] = (const float*)d_in[23 + 2 * i];
    tdb[i] = (const float*)d_in[24 + 2 * i];
  }
  const float* bnf_w = (const float*)d_in[29];
  const float* bnf_b = (const float*)d_in[30];
  const float* ws1   = (const float*)d_in[31];
  const float* ws2   = (const float*)d_in[32];
  const float* bnw_w = (const float*)d_in[33];
  const float* bnw_b = (const float*)d_in[34];

  float* out_f = (float*)d_out;              // (64,128)
  float* At    = out_f + 64 * 128;           // (64,5,782)

  // ---- workspace carve ----
  char* p = (char*)d_ws;
  auto carve = [&](size_t bytes) -> void* {
    void* r = (void*)p;
    p += (bytes + 255) & ~(size_t)255;
    return r;
  };
  const size_t actB = (size_t)B * T1 * H * 2;   // bf16 activation buffer
  unsigned short* A0 = (unsigned short*)carve(actB);
  unsigned short* A1 = (unsigned short*)carve(actB);
  unsigned short* A2 = (unsigned short*)carve(actB);
  unsigned short* w_fc1 = (unsigned short*)carve((size_t)H * D * 2);
  unsigned short *w_r1[4], *w_r2[4];
  for (int i = 0; i < 4; ++i) {
    w_r1[i] = (unsigned short*)carve((size_t)H * H * 2);
    w_r2[i] = (unsigned short*)carve((size_t)H * H * 2);
  }
  unsigned short *w_td0[3], *w_td1[3];
  for (int i = 0; i < 3; ++i) {
    w_td0[i] = (unsigned short*)carve((size_t)H * H * 2);
    w_td1[i] = (unsigned short*)carve((size_t)H * H * 2);
  }
  unsigned short* w_bnf = (unsigned short*)carve((size_t)128 * H * 2);
  unsigned short* w_s1  = (unsigned short*)carve((size_t)128 * 128 * 2);
  unsigned short* w_s2  = (unsigned short*)carve((size_t)5 * 128 * 2);
  float* scores = (float*)carve((size_t)B * T4 * 5 * 4);
  float* pooled = (float*)carve((size_t)B * 5 * 128 * 4);

  // ---- weight conversion ----
  auto cvt = [&](const float* s, unsigned short* d, long n) {
    k_f32_to_bf16<<<(unsigned)((n + 255) / 256), 256, 0, stream>>>(s, d, n);
  };
  cvt(fc1_w, w_fc1, (long)H * D);
  for (int i = 0; i < 4; ++i) {
    cvt(rw1[i], w_r1[i], (long)H * H);
    cvt(rw2[i], w_r2[i], (long)H * H);
  }
  for (int i = 0; i < 3; ++i)
    k_tdnn_split<<<(H * H + 255) / 256, 256, 0, stream>>>(tdk[i], w_td0[i], w_td1[i], H * H);
  cvt(bnf_w, w_bnf, (long)128 * H);
  cvt(ws1,   w_s1,  (long)128 * 128);
  cvt(ws2,   w_s2,  (long)5 * 128);

  dim3 blk(256);
  auto grid = [&](int N, int Tout) {
    return dim3((N + BN - 1) / BN, (Tout + BM - 1) / BM, B);
  };

  // 1) fc1: relu(x @ fc1_w^T + b)   f32 A converted on the fly
  gemm_wmma_bf16<<<grid(H, T1), blk, 0, stream>>>(x, 1, T1, T1, D, H, 0, 0, 1,
      w_fc1, nullptr, fc1_b, nullptr, nullptr, 1, A0, nullptr);

  int Tc = T1;
  unsigned short* h = A0;
  const int toff[3] = {3, 6, 9};
  const int tout[3] = {T2, T3, T4};

  for (int blkno = 0; blkno < 4; ++blkno) {
    // residual block on h (T = Tc)
    gemm_wmma_bf16<<<grid(H, Tc), blk, 0, stream>>>(h, 0, Tc, Tc, H, H, 0, 0, 1,
        w_r1[blkno], nullptr, rb1[blkno], nullptr, nullptr, 1, A1, nullptr);
    gemm_wmma_bf16<<<grid(H, Tc), blk, 0, stream>>>(A1, 0, Tc, Tc, H, H, 0, 0, 1,
        w_r2[blkno], nullptr, rb2[blkno], h, rbs[blkno], 1, A2, nullptr);
    h = A2;
    if (blkno < 3) {
      // tdnn layer blkno (2-tap): relu
      gemm_wmma_bf16<<<grid(H, tout[blkno]), blk, 0, stream>>>(h, 0, Tc, tout[blkno],
          H, H, 0, toff[blkno], 2, w_td0[blkno], w_td1[blkno], tdb[blkno],
          nullptr, nullptr, 1, A0, nullptr);
      Tc = tout[blkno];
      h = A0;
    }
  }

  // bnf: h @ bnf_w^T + b (no act) -> A0 (B,T4,128)
  gemm_wmma_bf16<<<grid(128, T4), blk, 0, stream>>>(h, 0, T4, T4, H, 128, 0, 0, 1,
      w_bnf, nullptr, bnf_b, nullptr, nullptr, 0, A0, nullptr);
  // e = tanh(h_bnf @ ws1^T) -> A1
  gemm_wmma_bf16<<<grid(128, T4), blk, 0, stream>>>(A0, 0, T4, T4, 128, 128, 0, 0, 1,
      w_s1, nullptr, nullptr, nullptr, nullptr, 2, A1, nullptr);
  // scores = e @ ws2^T -> f32 (B,T4,5)
  gemm_wmma_bf16<<<grid(5, T4), blk, 0, stream>>>(A1, 0, T4, T4, 128, 5, 0, 0, 1,
      w_s2, nullptr, nullptr, nullptr, nullptr, 0, nullptr, scores);

  // softmax over t -> At (B,5,T4) into d_out
  k_softmax_t<<<B * 5, 256, 0, stream>>>(scores, At, T4);
  // pooled (B,5,128)
  k_pool<<<dim3(5, B), 128, 0, stream>>>(At, A0, pooled, T4);
  // final out (B,128)
  k_final<<<B, 128, 0, stream>>>(pooled, bnw_w, bnw_b, out_f);
}
